// MS_MoE_Conv_10754598109427
// MI455X (gfx1250) — matrix operand order repaired and checked by reading, not compiled, not run on hardware
//
#include <hip/hip_runtime.h>
#include <hip/hip_bf16.h>

// ---------------------------------------------------------------------------
// MS-MoE-Conv forward for MI455X (gfx1250, wave32, WMMA f16->f32)
// ---------------------------------------------------------------------------

typedef __attribute__((ext_vector_type(16))) _Float16 v16h;
typedef __attribute__((ext_vector_type(8)))  float    v8f;

#define T_    4
#define B_    32
#define TB_   128          // T*B tokens
#define C_    256
#define HW_   196          // 14*14
#define E_    8
#define HID_  1024
#define SPK   264          // f16 stride of spike rows in LDS (bank-friendly: 132 dwords -> 4-bank lane shift)

// ---------------------------------------------------------------------------
// K0: convert fc1_w / fc2_w (fp32) to f16 in workspace (both are E*HID*C elems)
// ---------------------------------------------------------------------------
__global__ void convert_weights_kernel(const float* __restrict__ fc1,
                                       const float* __restrict__ fc2,
                                       _Float16* __restrict__ fc1h,
                                       _Float16* __restrict__ fc2h,
                                       int nele) {
    int i = blockIdx.x * blockDim.x + threadIdx.x;
    if (i < nele) {
        fc1h[i] = (_Float16)fc1[i];
        fc2h[i] = (_Float16)fc2[i];
    }
}

// ---------------------------------------------------------------------------
// K1: router LIF (tau=2, hard reset) scanned over T, mean over (h,w)
//     sbar[(t*B+b)*C + c] = mean_hw spike_t
// One block per (b,c); 256 threads (196 active pixels).
// ---------------------------------------------------------------------------
__global__ void lif_router_kernel(const float* __restrict__ x,
                                  float* __restrict__ sbar) {
    __shared__ float red[256];
    const int bc  = blockIdx.x;          // b*C + c
    const int b   = bc / C_;
    const int c   = bc % C_;
    const int tid = threadIdx.x;

    float s[T_];
    float v = 0.0f;
    if (tid < HW_) {
        #pragma unroll
        for (int t = 0; t < T_; ++t) {
            float xv = x[(((size_t)t * B_ + b) * C_ + c) * HW_ + tid];
            v = v + (xv - v) / 2.0f;                 // tau = 2
            float sp = (v - 1.0f >= 0.0f) ? 1.0f : 0.0f;
            v = v * (1.0f - sp);                     // hard reset
            s[t] = sp;
        }
    } else {
        #pragma unroll
        for (int t = 0; t < T_; ++t) s[t] = 0.0f;
    }

    for (int t = 0; t < T_; ++t) {
        red[tid] = s[t];
        __syncthreads();
        for (int off = 128; off > 0; off >>= 1) {
            if (tid < off) red[tid] += red[tid + off];
            __syncthreads();
        }
        if (tid == 0)
            sbar[((size_t)t * B_ + b) * C_ + c] = red[0] / (float)HW_;
        __syncthreads();
    }
}

// ---------------------------------------------------------------------------
// K2: router head: logits = (sbar @ Wr^T + br)*g + beta ; softmax ; top-2 ;
//     comb[tb,e] = normalized weight for selected experts, else 0.
// One block of 128 threads, one token per thread. Exact fp32.
// ---------------------------------------------------------------------------
__global__ void router_head_kernel(const float* __restrict__ sbar,
                                   const float* __restrict__ rw,
                                   const float* __restrict__ rb,
                                   const float* __restrict__ rg,
                                   const float* __restrict__ rbeta,
                                   float* __restrict__ comb) {
    const int row = threadIdx.x;
    if (row >= TB_) return;
    const float* sr = sbar + (size_t)row * C_;

    float logit[E_];
    #pragma unroll
    for (int e = 0; e < E_; ++e) {
        float acc = 0.0f;
        for (int c = 0; c < C_; ++c) acc += rw[e * C_ + c] * sr[c];
        logit[e] = (acc + rb[e]) * rg[e] + rbeta[e];
    }
    float mx = logit[0];
    #pragma unroll
    for (int e = 1; e < E_; ++e) mx = fmaxf(mx, logit[e]);
    float p[E_], sum = 0.0f;
    #pragma unroll
    for (int e = 0; e < E_; ++e) { p[e] = __expf(logit[e] - mx); sum += p[e]; }
    #pragma unroll
    for (int e = 0; e < E_; ++e) p[e] /= sum;

    int i1 = 0;
    for (int e = 1; e < E_; ++e) if (p[e] > p[i1]) i1 = e;        // first max on tie
    int i2 = (i1 == 0) ? 1 : 0;
    for (int e = 0; e < E_; ++e) if (e != i1 && p[e] > p[i2]) i2 = e;
    float wsum = p[i1] + p[i2];
    #pragma unroll
    for (int e = 0; e < E_; ++e) {
        float w = (e == i1) ? (p[i1] / wsum) : (e == i2) ? (p[i2] / wsum) : 0.0f;
        comb[row * E_ + e] = w;
    }
}

// ---------------------------------------------------------------------------
// K3: expert FFNs + combine + residual.
// Grid: TB blocks per pixel-split; block = 512 threads = 16 waves.
// Wave w owns output rows [w*16, w*16+16); WG owns p-tiles [TILE0, TILE0+NT).
// For each expert with comb != 0:
//   s1T (LDS, f16 0/1), then 4 chunks of 256 hidden rows:
//     GEMM1 (wmma f16) -> affine -> threshold -> s2T (LDS) -> GEMM2 (wmma f16)
// Accumulate affine(bn2)*comb into registers across experts; store out = x + acc.
// ---------------------------------------------------------------------------
template <int NT, int TILE0>
__global__ __launch_bounds__(512)
void moe_expert_kernel(const float*    __restrict__ x,
                       const _Float16* __restrict__ fc1h,
                       const _Float16* __restrict__ fc2h,
                       const float*    __restrict__ fc1_b,
                       const float*    __restrict__ bn1_g,
                       const float*    __restrict__ bn1_b,
                       const float*    __restrict__ fc2_b,
                       const float*    __restrict__ bn2_g,
                       const float*    __restrict__ bn2_b,
                       const float*    __restrict__ comb,
                       float*          __restrict__ out) {
    extern __shared__ _Float16 lds[];
    _Float16* s1T = lds;                          // [NT*16][SPK]  spikes of fc1 input  (cols = C)
    _Float16* s2T = lds + NT * 16 * SPK;          // [NT*16][SPK]  spikes of fc1 output (cols = chunk of HID)

    const int n    = blockIdx.x;                  // token
    const int tid  = threadIdx.x;
    const int w    = tid >> 5;                    // wave 0..15
    const int lane = tid & 31;
    const int nloc = lane & 15;                   // N index within tile / A-matrix M row
    const int half = lane >> 4;                   // lane half

    v8f out_acc[NT];
    #pragma unroll
    for (int t = 0; t < NT; ++t)
        #pragma unroll
        for (int r = 0; r < 8; ++r) out_acc[t][r] = 0.0f;

    const float* xn = x + (size_t)n * C_ * HW_;

    for (int e = 0; e < E_; ++e) {
        const float cw = comb[n * E_ + e];
        if (cw == 0.0f) continue;                 // uniform across block
        const float tau = (float)(1.9 + (double)e * (2.1 - 1.9) / (double)(E_ - 1));

        // ---- build s1T: s1[c,p] = (x >= tau), stored transposed [p][c] ----
        __syncthreads();
        for (int idx = tid; idx < C_ * NT * 16; idx += 512) {
            int c  = idx / (NT * 16);
            int pl = idx % (NT * 16);
            int pg = TILE0 * 16 + pl;
            float xv = (pg < HW_) ? xn[c * HW_ + pg] : 0.0f;
            s1T[pl * SPK + c] = (pg < HW_ && xv >= tau) ? (_Float16)1.0f : (_Float16)0.0f;
        }
        __syncthreads();

        v8f acc2[NT];
        #pragma unroll
        for (int t = 0; t < NT; ++t)
            #pragma unroll
            for (int r = 0; r < 8; ++r) acc2[t][r] = 0.0f;

        for (int fc = 0; fc < 4; ++fc) {          // 4 chunks of 256 hidden rows
            // per-M affine params of fc1/bn1 for this wave's rows
            float pfb[8], pgm[8], pbb[8];
            #pragma unroll
            for (int r = 0; r < 8; ++r) {
                int fg = fc * 256 + w * 16 + r + 8 * half;
                int o  = e * HID_ + fg;
                pfb[r] = fc1_b[o]; pgm[r] = bn1_g[o]; pbb[r] = bn1_b[o];
            }
            const _Float16* A1 =
                fc1h + (size_t)(e * HID_ + fc * 256 + w * 16 + nloc) * C_;

            // ---- GEMM1: h[f,p] = fc1_w @ s1 ; threshold -> s2T ----
            #pragma unroll
            for (int nt = 0; nt < NT; ++nt) {
                v8f acc;
                #pragma unroll
                for (int r = 0; r < 8; ++r) acc[r] = 0.0f;
                #pragma unroll
                for (int ks = 0; ks < 8; ++ks) {
                    const int k0 = ks * 32;
                    v16h a, b;
                    ((uint4*)&a)[0] = *(const uint4*)(A1 + k0 + 16 * half);
                    ((uint4*)&a)[1] = *(const uint4*)(A1 + k0 + 16 * half + 8);
                    const _Float16* brow = s1T + (nt * 16 + nloc) * SPK + k0 + 16 * half;
                    ((uint4*)&b)[0] = *(const uint4*)(brow);
                    ((uint4*)&b)[1] = *(const uint4*)(brow + 8);
                    acc = __builtin_amdgcn_wmma_f32_16x16x32_f16(
                              false, a, false, b, (short)0, acc, false, false);
                }
                #pragma unroll
                for (int r = 0; r < 8; ++r) {
                    float h = (acc[r] + pfb[r]) * pgm[r] + pbb[r];
                    int fl = w * 16 + r + 8 * half;
                    s2T[(nt * 16 + nloc) * SPK + fl] =
                        (h >= tau) ? (_Float16)1.0f : (_Float16)0.0f;
                }
            }
            __syncthreads();

            // ---- GEMM2 partial: o[c,p] += fc2_w[:, chunk] @ s2 ----
            const _Float16* A2 =
                fc2h + (size_t)(e * C_ + w * 16 + nloc) * HID_ + fc * 256;
            #pragma unroll
            for (int nt = 0; nt < NT; ++nt) {
                #pragma unroll
                for (int ks = 0; ks < 8; ++ks) {
                    const int k0 = ks * 32;
                    v16h a, b;
                    ((uint4*)&a)[0] = *(const uint4*)(A2 + k0 + 16 * half);
                    ((uint4*)&a)[1] = *(const uint4*)(A2 + k0 + 16 * half + 8);
                    const _Float16* brow = s2T + (nt * 16 + nloc) * SPK + k0 + 16 * half;
                    ((uint4*)&b)[0] = *(const uint4*)(brow);
                    ((uint4*)&b)[1] = *(const uint4*)(brow + 8);
                    acc2[nt] = __builtin_amdgcn_wmma_f32_16x16x32_f16(
                                   false, a, false, b, (short)0, acc2[nt], false, false);
                }
            }
            __syncthreads();                      // before s2T is overwritten
        }

        // ---- bn2 affine, weight by comb, accumulate across experts ----
        #pragma unroll
        for (int nt = 0; nt < NT; ++nt) {
            #pragma unroll
            for (int r = 0; r < 8; ++r) {
                int c = w * 16 + r + 8 * half;
                int o = e * C_ + c;
                float v = (acc2[nt][r] + fc2_b[o]) * bn2_g[o] + bn2_b[o];
                out_acc[nt][r] += v * cw;
            }
        }
    }

    // ---- out = x + sum_e comb*o  (each (n,c,p) written exactly once) ----
    #pragma unroll
    for (int nt = 0; nt < NT; ++nt) {
        int pg = (TILE0 + nt) * 16 + nloc;
        if (pg < HW_) {
            #pragma unroll
            for (int r = 0; r < 8; ++r) {
                int c = w * 16 + r + 8 * half;
                size_t oi = (size_t)n * C_ * HW_ + (size_t)c * HW_ + pg;
                out[oi] = xn[c * HW_ + pg] + out_acc[nt][r];
            }
        }
    }
}

// ---------------------------------------------------------------------------
// Host-side launch
// ---------------------------------------------------------------------------
extern "C" void kernel_launch(void* const* d_in, const int* in_sizes, int n_in,
                              void* d_out, int out_size, void* d_ws, size_t ws_size,
                              hipStream_t stream) {
    const float* x           = (const float*)d_in[0];
    const float* router_w    = (const float*)d_in[1];
    const float* router_b    = (const float*)d_in[2];
    const float* router_g    = (const float*)d_in[3];
    const float* router_beta = (const float*)d_in[4];
    const float* fc1_w       = (const float*)d_in[5];
    const float* fc1_b       = (const float*)d_in[6];
    const float* bn1_g       = (const float*)d_in[7];
    const float* bn1_b       = (const float*)d_in[8];
    const float* fc2_w       = (const float*)d_in[9];
    const float* fc2_b       = (const float*)d_in[10];
    const float* bn2_g       = (const float*)d_in[11];
    const float* bn2_b       = (const float*)d_in[12];
    float* out = (float*)d_out;

    char* ws = (char*)d_ws;
    _Float16* fc1h = (_Float16*)(ws);                      // 4 MB
    _Float16* fc2h = (_Float16*)(ws + 4194304);            // 4 MB
    float*    sbar = (float*)(ws + 8388608);               // 128*256*4
    float*    comb = (float*)(ws + 8519680);               // 128*8*4

    // K0: fp32 -> f16 weight conversion
    {
        int nele = E_ * HID_ * C_;                         // 2,097,152 (same for both)
        convert_weights_kernel<<<nele / 256, 256, 0, stream>>>(fc1_w, fc2_w, fc1h, fc2h, nele);
    }
    // K1: router LIF + spatial mean
    lif_router_kernel<<<B_ * C_, 256, 0, stream>>>(x, sbar);
    // K2: logits/softmax/top-2 -> comb
    router_head_kernel<<<1, 128, 0, stream>>>(sbar, router_w, router_b,
                                              router_g, router_beta, comb);
    // K3: expert FFNs (pixel tiles 0..6 and 7..12)
    {
        size_t sh7 = (size_t)2 * 7 * 16 * SPK * sizeof(_Float16);   // 118,272 B
        size_t sh6 = (size_t)2 * 6 * 16 * SPK * sizeof(_Float16);   // 101,376 B
        moe_expert_kernel<7, 0><<<TB_, 512, sh7, stream>>>(
            x, fc1h, fc2h, fc1_b, bn1_g, bn1_b, fc2_b, bn2_g, bn2_b, comb, out);
        moe_expert_kernel<6, 7><<<TB_, 512, sh6, stream>>>(
            x, fc1h, fc2h, fc1_b, bn1_g, bn1_b, fc2_b, bn2_g, bn2_b, comb, out);
    }
}
